// DotProductAttention_47863115547534
// MI455X (gfx1250) — matrix operand order repaired and checked by reading, not compiled, not run on hardware
//
#include <hip/hip_runtime.h>
#include <hip/hip_bf16.h>

typedef __attribute__((ext_vector_type(16))) _Float16 v16h;
typedef __attribute__((ext_vector_type(8)))  _Float16 v8h;
typedef __attribute__((ext_vector_type(8)))  float    v8f;

#define B_ 4
#define H_ 8
#define S_ 2048
#define D_ 256
#define SCALE_INV 0.0625f
#define LOG2E 1.4426950408889634f

#define QTILE 128       // query rows per workgroup (8 waves x 16)
#define KCHUNK 32       // keys per flash iteration
#define NWAVE 8

#define KLDS_STRIDE 264 // 256 + 8 halves pad (row stride 528B: 16B aligned, bank-skewed)
#define VT_STRIDE   40  // 32 + 8 halves pad (row stride 80B: 16B aligned, bank-skewed)
#define P_STRIDE    40  // 32 + 8 halves pad

static __device__ __forceinline__ uint32_t pkhalf2(float x, float y) {
    union { _Float16 h[2]; uint32_t u; } t;
    t.h[0] = (_Float16)x; t.h[1] = (_Float16)y;
    return t.u;
}

static __device__ __forceinline__ v16h cat16(v8h lo, v8h hi) {
    v16h r;
#pragma unroll
    for (int j = 0; j < 8; ++j) { r[j] = lo[j]; r[8 + j] = hi[j]; }
    return r;
}

// 16 contiguous halves from LDS (16B-aligned) as two v8h reads -> ds_load_b128 x2
static __device__ __forceinline__ v16h ldsv16(const _Float16* p) {
    v8h lo = *(const v8h*)p;
    v8h hi = *(const v8h*)(p + 8);
    return cat16(lo, hi);
}

__global__ __launch_bounds__(256, 1)
void fa_fwd_kernel(const float* __restrict__ Qg, const float* __restrict__ Kg,
                   const float* __restrict__ Vg, float* __restrict__ Og)
{
    __shared__ _Float16 Klds[KCHUNK * KLDS_STRIDE];      // K chunk, row-major [key][d]
    __shared__ _Float16 Vtlds[D_ * VT_STRIDE];           // V chunk, transposed [d][key]
    __shared__ _Float16 Plds[NWAVE * 16 * P_STRIDE];     // per-wave P tile [M][k]

    const int tid  = threadIdx.x;
    const int lane = tid & 31;
    const int n    = lane & 15;   // column / row-within-16 index
    const int h    = lane >> 4;   // lane half
    const int w    = tid >> 5;    // wave id (0..7)

    const int bh  = blockIdx.y;
    const int qw  = blockIdx.x * QTILE + w * 16;   // this wave's first query row

    const size_t base = (size_t)bh * S_ * D_;
    const float* Qp = Qg + base;
    const float* Kp = Kg + base;
    const float* Vp = Vg + base;
    float*       Op = Og + base;

    // ---- Q -> f16 A-fragments for all 8 K-chunks of D=256 (lane row M = lane%16) ----
    v16h aQ[8];
    {
        const float* qrow = Qp + (size_t)(qw + n) * D_;
#pragma unroll
        for (int c = 0; c < 8; ++c) {
            v16h a;
#pragma unroll
            for (int i = 0; i < 8; ++i) {
                // contiguous A layout: VGPR i holds K = 16h + 2*i, +1
                const int d = 32 * c + 16 * h + 2 * i;
                const float2 q2 = *(const float2*)(qrow + d);
                a[2 * i]     = (_Float16)q2.x;
                a[2 * i + 1] = (_Float16)q2.y;
            }
            aQ[c] = a;
        }
    }

    const v8f zero = {};
    v8f acc[16];                 // 16 x (16x16 f32 C tiles) covering D=256
#pragma unroll
    for (int t = 0; t < 16; ++t) acc[t] = zero;

    float m[8], l[8];
#pragma unroll
    for (int p = 0; p < 8; ++p) { m[p] = -3.0e38f; l[p] = 0.0f; }

    _Float16* Pw = &Plds[w * 16 * P_STRIDE];

    const int NJ = S_ / KCHUNK; // 64
    for (int j = 0; j < NJ; ++j) {
        const int jb = j * KCHUNK;

        __syncthreads();  // previous iteration's LDS reads complete
        // ---- stage K (row-major f16) and V (transposed f16), 256 threads, packed b32 stores ----
#pragma unroll 4
        for (int it = 0; it < 16; ++it) {
            const int idx = it * 256 + tid;
            // K: key = idx/128, d-pair = (idx%128)*2
            {
                const int key = idx >> 7;
                const int dp  = (idx & 127) * 2;
                const float2 k2 = *(const float2*)(Kp + (size_t)(jb + key) * D_ + dp);
                *(uint32_t*)&Klds[key * KLDS_STRIDE + dp] = pkhalf2(k2.x, k2.y);
            }
            // V: d = idx%256, key-pair = idx/256 -> Vt[d][2kp],Vt[d][2kp+1]
            {
                const int d  = idx & 255;
                const int kp = idx >> 8;
                const float a0 = Vp[(size_t)(jb + 2 * kp    ) * D_ + d];
                const float a1 = Vp[(size_t)(jb + 2 * kp + 1) * D_ + d];
                *(uint32_t*)&Vtlds[d * VT_STRIDE + 2 * kp] = pkhalf2(a0, a1);
            }
        }
        if (j + 1 < NJ) {  // hint next chunk into cache (global_prefetch_b8)
            __builtin_prefetch(Kp + (size_t)(jb + KCHUNK) * D_ + (size_t)tid * 32, 0, 1);
            __builtin_prefetch(Vp + (size_t)(jb + KCHUNK) * D_ + (size_t)tid * 32, 0, 1);
        }
        __syncthreads();

        // ---- scores: S[16q x 32k] = Q.K^T, two 16x16 tiles, 8 d-chunks each ----
        v8f sc0 = zero, sc1 = zero;
#pragma unroll
        for (int c = 0; c < 8; ++c) {
            // B-fragment: lane col = n (key), VGPR i holds d-K = 16h + 2i -> contiguous halves
            const v16h kb0 = ldsv16(&Klds[(n     ) * KLDS_STRIDE + 32 * c + 16 * h]);
            const v16h kb1 = ldsv16(&Klds[(16 + n) * KLDS_STRIDE + 32 * c + 16 * h]);
            sc0 = __builtin_amdgcn_wmma_f32_16x16x32_f16(false, aQ[c], false, kb0, (short)0, sc0, false, false);
            sc1 = __builtin_amdgcn_wmma_f32_16x16x32_f16(false, aQ[c], false, kb1, (short)0, sc1, false, false);
        }

        // ---- online softmax over the 32 keys; rows live across 16 lanes of one half ----
        float alpha[8];
#pragma unroll
        for (int p = 0; p < 8; ++p) {
            float s0 = sc0[p] * SCALE_INV;
            float s1 = sc1[p] * SCALE_INV;
            float rm = fmaxf(s0, s1);
            rm = fmaxf(rm, __shfl_xor(rm, 1, 32));
            rm = fmaxf(rm, __shfl_xor(rm, 2, 32));
            rm = fmaxf(rm, __shfl_xor(rm, 4, 32));
            rm = fmaxf(rm, __shfl_xor(rm, 8, 32));
            const float mn = fmaxf(m[p], rm);
            const float e0 = __builtin_amdgcn_exp2f((s0 - mn) * LOG2E);
            const float e1 = __builtin_amdgcn_exp2f((s1 - mn) * LOG2E);
            float rs = e0 + e1;
            rs += __shfl_xor(rs, 1, 32);
            rs += __shfl_xor(rs, 2, 32);
            rs += __shfl_xor(rs, 4, 32);
            rs += __shfl_xor(rs, 8, 32);
            const float al = __builtin_amdgcn_exp2f((m[p] - mn) * LOG2E);
            l[p] = l[p] * al + rs;
            m[p] = mn;
            alpha[p] = al;
            // C-layout -> row-major P in wave-private LDS (row M = p + 8h, col = n / 16+n)
            Pw[(p + 8 * h) * P_STRIDE + n]      = (_Float16)e0;
            Pw[(p + 8 * h) * P_STRIDE + 16 + n] = (_Float16)e1;
        }

        // ---- reload P as an A-fragment: lane row M = n; K runs [16h..16h+15] ----
        const v16h pa = cat16(*(const v8h*)&Pw[n * P_STRIDE + 16 * h],
                              *(const v8h*)&Pw[n * P_STRIDE + 16 * h + 8]);

        // ---- O += P.V : 16 d-tiles; rescale accumulators by alpha first ----
#pragma unroll
        for (int t = 0; t < 16; ++t) {
#pragma unroll
            for (int p = 0; p < 8; ++p) acc[t][p] *= alpha[p];
            // B-fragment: lane col = d (16t+n), VGPR i holds key-K = 16h + 2i -> contiguous Vt row
            const v16h vb = ldsv16(&Vtlds[(16 * t + n) * VT_STRIDE + 16 * h]);
            acc[t] = __builtin_amdgcn_wmma_f32_16x16x32_f16(false, pa, false, vb, (short)0, acc[t], false, false);
        }
    }

    // ---- normalize and store; out flat layout == [B,H,S,D] ----
    float inv[8];
#pragma unroll
    for (int p = 0; p < 8; ++p) inv[p] = 1.0f / l[p];
#pragma unroll
    for (int t = 0; t < 16; ++t) {
#pragma unroll
        for (int p = 0; p < 8; ++p) {
            const int q = qw + p + 8 * h;
            Op[(size_t)q * D_ + 16 * t + n] = acc[t][p] * inv[p];
        }
    }
}

extern "C" void kernel_launch(void* const* d_in, const int* in_sizes, int n_in,
                              void* d_out, int out_size, void* d_ws, size_t ws_size,
                              hipStream_t stream) {
    (void)in_sizes; (void)n_in; (void)out_size; (void)d_ws; (void)ws_size;
    const float* Q = (const float*)d_in[0];
    const float* K = (const float*)d_in[1];
    const float* V = (const float*)d_in[2];
    float* O = (float*)d_out;

    dim3 grid(S_ / QTILE, B_ * H_);   // (16, 32)
    dim3 block(256);                   // 8 waves (wave32)
    fa_fwd_kernel<<<grid, block, 0, stream>>>(Q, K, V, O);
}
